// FP8QDQLinear_65575560675871
// MI455X (gfx1250) — 8-bit matrix operand order repaired and checked by reading, compiled, not run on hardware
//
#include <hip/hip_runtime.h>


typedef __attribute__((ext_vector_type(16))) int          v16i;
typedef __attribute__((ext_vector_type(8)))  float        v8f;
typedef __attribute__((ext_vector_type(4)))  int          v4i;
typedef __attribute__((ext_vector_type(4)))  float        v4f;
typedef __attribute__((ext_vector_type(2)))  unsigned int v2u;

#define TILE_M 128
#define TILE_N 128
#define TILE_K 128
#define KPITCH 144
#define A_PITCH 136
#define B_PITCH 144

__device__ __forceinline__ float qprep(float x) {
  x *= 0.5f;
  return fminf(448.0f, fmaxf(-448.0f, x));
}

#if defined(__has_builtin)
#if __has_builtin(__builtin_amdgcn_cvt_pk_fp8_f32)
#define HAVE_HW_FP8_CVT 1
#endif
#if __has_builtin(__builtin_amdgcn_global_load_async_to_lds_b128)
#define HAVE_ASYNC_LDS 1
#endif
#endif

#ifndef HAVE_HW_FP8_CVT
__device__ __forceinline__ unsigned int f32_to_e4m3(float f) {
  unsigned int u = __float_as_uint(f);
  unsigned int s = (u >> 24) & 0x80u;
  int e = (int)((u >> 23) & 0xFFu) - 127;
  unsigned int m = u & 0x7FFFFFu;
  if (e < -9) return s;
  if (e >= -6) {
    unsigned int mant = m >> 20;
    unsigned int rem  = m & 0xFFFFFu;
    mant += (rem > 0x80000u) || (rem == 0x80000u && (mant & 1u));
    unsigned int val = ((unsigned int)(e + 7) << 3) + mant;
    if (val > 0x7Eu) val = 0x7Eu;
    return s | val;
  }
  unsigned int full = m | 0x800000u;
  int shift = 20 + (-6 - e);
  unsigned int mant = full >> shift;
  unsigned int rem  = full & ((1u << shift) - 1u);
  unsigned int half = 1u << (shift - 1);
  mant += (rem > half) || (rem == half && (mant & 1u));
  return s | mant;
}
#endif

__device__ __forceinline__ int pack4_fp8(float a, float b, float c, float d) {
#ifdef HAVE_HW_FP8_CVT
  int p = 0;
  p = __builtin_amdgcn_cvt_pk_fp8_f32(a, b, p, false);
  p = __builtin_amdgcn_cvt_pk_fp8_f32(c, d, p, true);
  return p;
#else
  return (int)(f32_to_e4m3(a) | (f32_to_e4m3(b) << 8) |
               (f32_to_e4m3(c) << 16) | (f32_to_e4m3(d) << 24));
#endif
}

#ifdef HAVE_ASYNC_LDS
__device__ __forceinline__ void async_copy_b128(const void* g, void* lds_generic) {
  __builtin_amdgcn_global_load_async_to_lds_b128((v4i*)g, (v4i*)lds_generic, 0, 0);
}
#if __has_builtin(__builtin_amdgcn_s_wait_asynccnt)
#define WAIT_ASYNC() __builtin_amdgcn_s_wait_asynccnt(0)
#else
#define WAIT_ASYNC() asm volatile("s_wait_asynccnt 0x0" ::: "memory")
#endif
#else
__device__ __forceinline__ void async_copy_b128(const void* g, void* lds_generic) {
  *(v4i*)lds_generic = *(const v4i*)g;
}
#define WAIT_ASYNC() ((void)0)
#endif

__device__ __forceinline__ void stage_tile(const unsigned char* gbase, size_t gstride,
                                           unsigned char* lbase, int tid) {
#pragma unroll
  for (int i = 0; i < 4; ++i) {
    const int idx = i * 256 + tid;
    const int row = idx >> 3;
    const int c16 = (idx & 7) * 16;
    async_copy_b128(gbase + (size_t)row * gstride + c16,
                    lbase + row * KPITCH + c16);
  }
}

__launch_bounds__(256, 4)
__global__ void fp8_quantize_kernel(const float* __restrict__ X,
                                    unsigned char* __restrict__ Xq,
                                    long long n16) {
  const long long t = (long long)blockIdx.x * 256 + threadIdx.x;
  if (t >= n16) return;
  const float* src = X + t * 16;
  v4i out;
#pragma unroll
  for (int j = 0; j < 4; ++j) {
    const v4f f = *(const v4f*)(src + j * 4);
    out[j] = pack4_fp8(qprep(f.x), qprep(f.y), qprep(f.z), qprep(f.w));
  }
  *(v4i*)(Xq + t * 16) = out;
}

__launch_bounds__(256, 1)
__global__ void fp8_gemm_kernel(const unsigned char* __restrict__ Aq,
                                const unsigned char* __restrict__ W,
                                const float* __restrict__ bias,
                                float* __restrict__ Y,
                                int M, int N, int K) {
  __shared__ unsigned char sA[2][TILE_M * KPITCH];
  __shared__ unsigned char sB[2][TILE_N * KPITCH];

  const int tid  = threadIdx.x;
  const int lane = tid & 31;
  const int wave = tid >> 5;
  const int l16  = lane & 15;
  const int lhi  = lane >> 4;
  const int wm   = wave >> 1;
  const int wn   = wave & 1;

  const int M0 = blockIdx.y * TILE_M;
  const int N0 = blockIdx.x * TILE_N;

  const unsigned char* Abase = Aq + (size_t)M0 * K;
  const unsigned char* Wbase = W  + (size_t)N0 * K;

  const v8f vzero = {0.f, 0.f, 0.f, 0.f, 0.f, 0.f, 0.f, 0.f};
  v8f acc[2][4];
#pragma unroll
  for (int mt = 0; mt < 2; ++mt)
#pragma unroll
    for (int nt = 0; nt < 4; ++nt) acc[mt][nt] = vzero;

  stage_tile(Abase, K, &sA[0][0], tid);
  stage_tile(Wbase, K, &sB[0][0], tid);

  const int KSTEPS = K / TILE_K;
  for (int ks = 0; ks < KSTEPS; ++ks) {
    const int cur = ks & 1;
    WAIT_ASYNC();
    __syncthreads();

    if (ks + 1 < KSTEPS) {
      const int nxt = cur ^ 1;
      stage_tile(Abase + (ks + 1) * TILE_K, K, &sA[nxt][0], tid);
      stage_tile(Wbase + (ks + 1) * TILE_K, K, &sB[nxt][0], tid);
    }

    v16i afr[2];
#pragma unroll
    for (int mt = 0; mt < 2; ++mt) {
      const unsigned char* ap =
          &sA[cur][0] + (wm * 32 + mt * 16 + l16) * KPITCH + lhi * 16;
#pragma unroll
      for (int j = 0; j < 8; ++j) {
        const v2u t = *(const v2u*)(ap + (j >> 1) * 32 + (j & 1) * 8);
        afr[mt][2 * j]     = (int)t.x;
        afr[mt][2 * j + 1] = (int)t.y;
      }
    }
    v16i bfr[4];
#pragma unroll
    for (int nt = 0; nt < 4; ++nt) {
      const unsigned char* bp =
          &sB[cur][0] + (wn * 64 + nt * 16 + l16) * KPITCH + lhi * 16;
#pragma unroll
      for (int j = 0; j < 4; ++j) {
        const v4i t = *(const v4i*)(bp + j * 32);
        bfr[nt][4 * j]     = t.x;
        bfr[nt][4 * j + 1] = t.y;
        bfr[nt][4 * j + 2] = t.z;
        bfr[nt][4 * j + 3] = t.w;
      }
    }

#pragma unroll
    for (int mt = 0; mt < 2; ++mt)
#pragma unroll
      for (int nt = 0; nt < 4; ++nt)
        acc[mt][nt] = __builtin_amdgcn_wmma_f32_16x16x128_fp8_fp8(
            afr[mt], bfr[nt], (short)0, acc[mt][nt], false, false);

    __syncthreads();
  }

#pragma unroll
  for (int nt = 0; nt < 4; ++nt) {
    const int col = N0 + wn * 64 + nt * 16 + l16;
    const float bv = bias[col];
#pragma unroll
    for (int mt = 0; mt < 2; ++mt) {
      const int rowb = M0 + wm * 32 + mt * 16 + lhi * 8;
#pragma unroll
      for (int v = 0; v < 8; ++v)
        Y[(size_t)(rowb + v) * N + col] = acc[mt][nt][v] * 4.0f + bv;
    }
  }
}

__launch_bounds__(256, 1)
__global__ void fp8_qdq_linear_fused_kernel(const float* __restrict__ X,
                                            const unsigned char* __restrict__ W,
                                            const float* __restrict__ bias,
                                            float* __restrict__ Y,
                                            int M, int N, int K) {
  __shared__ unsigned char sA[TILE_M * A_PITCH];
  __shared__ unsigned char sB[TILE_N * B_PITCH];

  const int tid  = threadIdx.x;
  const int lane = tid & 31;
  const int wave = tid >> 5;
  const int l16  = lane & 15;
  const int lhi  = lane >> 4;
  const int wm   = wave >> 1;
  const int wn   = wave & 1;

  const int M0 = blockIdx.y * TILE_M;
  const int N0 = blockIdx.x * TILE_N;

  const v8f vzero = {0.f, 0.f, 0.f, 0.f, 0.f, 0.f, 0.f, 0.f};
  v8f acc[2][4];
#pragma unroll
  for (int mt = 0; mt < 2; ++mt)
#pragma unroll
    for (int nt = 0; nt < 4; ++nt) acc[mt][nt] = vzero;

  for (int k0 = 0; k0 < K; k0 += TILE_K) {
#pragma unroll
    for (int i = 0; i < 16; ++i) {
      const int idx = i * 256 + tid;
      const int row = idx >> 5;
      const int c4  = idx & 31;
      const v4f f = *(const v4f*)(X + (size_t)(M0 + row) * K + k0 + c4 * 4);
      *(int*)(sA + row * A_PITCH + c4 * 4) =
          pack4_fp8(qprep(f.x), qprep(f.y), qprep(f.z), qprep(f.w));
    }
#pragma unroll
    for (int i = 0; i < 4; ++i) {
      const int idx = i * 256 + tid;
      const int row = idx >> 3;
      const int c16 = idx & 7;
      const size_t goff = (size_t)(N0 + row) * K + k0 + c16 * 16;
      *(v4i*)(sB + row * B_PITCH + c16 * 16) = *(const v4i*)(W + goff);
      if (k0 + TILE_K < K) __builtin_prefetch(W + goff + TILE_K, 0, 0);
    }
    __syncthreads();

    v16i afr[2];
#pragma unroll
    for (int mt = 0; mt < 2; ++mt) {
      const unsigned char* ap = sA + (wm * 32 + mt * 16 + l16) * A_PITCH + lhi * 16;
#pragma unroll
      for (int j = 0; j < 8; ++j) {
        const v2u t = *(const v2u*)(ap + (j >> 1) * 32 + (j & 1) * 8);
        afr[mt][2 * j]     = (int)t.x;
        afr[mt][2 * j + 1] = (int)t.y;
      }
    }
    v16i bfr[4];
#pragma unroll
    for (int nt = 0; nt < 4; ++nt) {
      const unsigned char* bp = sB + (wn * 64 + nt * 16 + l16) * B_PITCH + lhi * 16;
#pragma unroll
      for (int j = 0; j < 4; ++j) {
        const v4i t = *(const v4i*)(bp + j * 32);
        bfr[nt][4 * j]     = t.x;
        bfr[nt][4 * j + 1] = t.y;
        bfr[nt][4 * j + 2] = t.z;
        bfr[nt][4 * j + 3] = t.w;
      }
    }
#pragma unroll
    for (int mt = 0; mt < 2; ++mt)
#pragma unroll
      for (int nt = 0; nt < 4; ++nt)
        acc[mt][nt] = __builtin_amdgcn_wmma_f32_16x16x128_fp8_fp8(
            afr[mt], bfr[nt], (short)0, acc[mt][nt], false, false);
    __syncthreads();
  }

#pragma unroll
  for (int nt = 0; nt < 4; ++nt) {
    const int col = N0 + wn * 64 + nt * 16 + l16;
    const float bv = bias[col];
#pragma unroll
    for (int mt = 0; mt < 2; ++mt) {
      const int rowb = M0 + wm * 32 + mt * 16 + lhi * 8;
#pragma unroll
      for (int v = 0; v < 8; ++v)
        Y[(size_t)(rowb + v) * N + col] = acc[mt][nt][v] * 4.0f + bv;
    }
  }
}

extern "C" void kernel_launch(void* const* d_in, const int* in_sizes, int n_in,
                              void* d_out, int out_size, void* d_ws, size_t ws_size,
                              hipStream_t stream) {
  const float*         X    = (const float*)d_in[0];
  const unsigned char* W    = (const unsigned char*)d_in[1];
  const float*         bias = (const float*)d_in[2];
  float*               Y    = (float*)d_out;

  const int N = in_sizes[2];
  const int K = in_sizes[1] / N;
  const int M = in_sizes[0] / K;

  const dim3 grid(N / TILE_N, M / TILE_M);
  const size_t needA = (size_t)M * (size_t)K;

  if (ws_size >= needA) {
    unsigned char* Xq = (unsigned char*)d_ws;
    const long long n16 = (long long)M * K / 16;
    const unsigned nblk = (unsigned)((n16 + 255) / 256);
    fp8_quantize_kernel<<<dim3(nblk), dim3(256), 0, stream>>>(X, Xq, n16);
    fp8_gemm_kernel<<<grid, dim3(256), 0, stream>>>(Xq, W, bias, Y, M, N, K);
  } else {
    fp8_qdq_linear_fused_kernel<<<grid, dim3(256), 0, stream>>>(X, W, bias, Y, M, N, K);
  }
}
